// ModelTop2Gate_69741678952704
// MI455X (gfx1250) — compile-verified, matrix operand order checked by reading, not run on hardware
//
#include <hip/hip_runtime.h>
#include <hip/hip_bf16.h>

typedef __attribute__((ext_vector_type(16))) _Float16 v16h;
typedef __attribute__((ext_vector_type(8)))  float    v8f;

#define S_TOK 4096
#define N_EXP 64
#define NEG_FLT_MAX (-3.402823466e+38f)
#define FLT_EPS 1.1920928955078125e-07f

// ---------------------------------------------------------------------------
// K1: one wave32 per token. Softmax over 64 experts (2 per lane), top-1
// argmax of gates, top-2 argmax of (logits+noise) with top-1 masked out.
// ---------------------------------------------------------------------------
__global__ void gate_kernel(const float* __restrict__ logits,
                            const float* __restrict__ noise,
                            float* __restrict__ gates,
                            int* __restrict__ idx1, int* __restrict__ idx2,
                            float* __restrict__ g1raw, float* __restrict__ g2raw) {
    const int wave = threadIdx.x >> 5;
    const int lane = threadIdx.x & 31;
    const int s = blockIdx.x * 8 + wave;                  // blockDim = 256 -> 8 tokens/block

    const size_t row = (size_t)s * N_EXP;
    float l0 = logits[row + lane];
    float l1 = logits[row + lane + 32];

    // softmax over 64
    float m = fmaxf(l0, l1);
    for (int o = 16; o > 0; o >>= 1) m = fmaxf(m, __shfl_xor(m, o, 32));
    float e0 = __expf(l0 - m), e1 = __expf(l1 - m);
    float ssum = e0 + e1;
    for (int o = 16; o > 0; o >>= 1) ssum += __shfl_xor(ssum, o, 32);
    float inv = 1.0f / ssum;
    float gg0 = e0 * inv, gg1 = e1 * inv;
    gates[row + lane]      = gg0;
    gates[row + lane + 32] = gg1;

    // top-1 argmax of gates (ties -> lowest index)
    float bv; int bi;
    if (gg0 >= gg1) { bv = gg0; bi = lane; } else { bv = gg1; bi = lane + 32; }
    for (int o = 16; o > 0; o >>= 1) {
        float ov = __shfl_xor(bv, o, 32);
        int   oi = __shfl_xor(bi, o, 32);
        if (ov > bv || (ov == bv && oi < bi)) { bv = ov; bi = oi; }
    }
    const int i1 = bi; const float gv1 = bv;

    // top-2: argmax of noisy logits with top-1 masked to -FLT_MAX; carry gate value
    float w0 = l0 + noise[row + lane];
    float w1 = l1 + noise[row + lane + 32];
    if (lane == i1)      w0 = NEG_FLT_MAX;
    if (lane + 32 == i1) w1 = NEG_FLT_MAX;
    float cv; int ci; float cg;
    if (w0 >= w1) { cv = w0; ci = lane; cg = gg0; } else { cv = w1; ci = lane + 32; cg = gg1; }
    for (int o = 16; o > 0; o >>= 1) {
        float ov = __shfl_xor(cv, o, 32);
        int   oi = __shfl_xor(ci, o, 32);
        float og = __shfl_xor(cg, o, 32);
        if (ov > cv || (ov == cv && oi < ci)) { cv = ov; ci = oi; cg = og; }
    }
    if (lane == 0) { idx1[s] = i1; idx2[s] = ci; g1raw[s] = gv1; g2raw[s] = cg; }
}

// ---------------------------------------------------------------------------
// K2: per-expert ordered rank (cumsum of one-hot over tokens) via wave32
// ballot prefix counting. 64 blocks x 1 wave.
// ---------------------------------------------------------------------------
__global__ void scan_kernel(const int* __restrict__ idx1, const int* __restrict__ idx2,
                            int* __restrict__ loc1, int* __restrict__ loc2,
                            int* __restrict__ cnt1) {
    const int e = blockIdx.x;
    const int lane = threadIdx.x;                          // blockDim = 32
    const unsigned lt = (1u << lane) - 1u;

    int base = 0;
    for (int t = 0; t < S_TOK; t += 32) {
        int i = idx1[t + lane];
        unsigned mask = __builtin_amdgcn_ballot_w32(i == e);
        if (i == e) loc1[t + lane] = base + __popc(mask & lt);
        base += __popc(mask);
    }
    if (lane == 0) cnt1[e] = base;

    int base2 = base;                                      // locations2 += sum(mask1) per expert
    for (int t = 0; t < S_TOK; t += 32) {
        int i = idx2[t + lane];
        unsigned mask = __builtin_amdgcn_ballot_w32(i == e);
        if (i == e) loc2[t + lane] = base2 + __popc(mask & lt);
        base2 += __popc(mask);
    }
}

// ---------------------------------------------------------------------------
// K3: l_aux. Column sums of gates via v_wmma_f32_16x16x32_f16 with an
// all-ones A matrix (D[m,n] = sum_k B[k,n]). 8 waves: 4 expert tiles x
// 2 token halves; exactly two LDS ds_add_f32 contributors per expert slot
// (order-invariant). Loop uses a LITERAL trip count so the compiler emits a
// uniform (scalar-branch) loop -> EXEC is structurally all-ones at each WMMA.
// l_aux = E * sum_e me_sum[e]*cnt1[e] / S^2.
// ---------------------------------------------------------------------------
__global__ void laux_kernel(const float* __restrict__ gates,
                            const int* __restrict__ cnt1,
                            float* __restrict__ out_laux) {
    __shared__ float me_sum[N_EXP];
    __shared__ float red[2];
    const int tid  = threadIdx.x;                          // blockDim = 256
    const int wave = tid >> 5;
    const int lane = tid & 31;
    if (tid < N_EXP) me_sum[tid] = 0.0f;
    __syncthreads();

    const int tile = wave & 3;                             // expert tile (16 experts)
    const int half = wave >> 2;                            // token half (2048 tokens)
    const int eb  = tile * 16;
    const int n   = eb + (lane & 15);
    const int k0  = (lane < 16) ? 0 : 8;
    const int tb0 = half * 2048;

    v16h a;
#pragma unroll
    for (int j = 0; j < 16; ++j) a[j] = (_Float16)1.0f;

    v8f acc = {};
    for (int it = 0; it < 64; ++it) {                      // literal bound: uniform loop
        const int tb = tb0 + it * 32;
        v16h b;
#pragma unroll
        for (int j = 0; j < 8; ++j)
            b[j]     = (_Float16)gates[(size_t)(tb + k0 + j) * N_EXP + n];
#pragma unroll
        for (int j = 0; j < 8; ++j)
            b[8 + j] = (_Float16)gates[(size_t)(tb + 16 + k0 + j) * N_EXP + n];
        acc = __builtin_amdgcn_wmma_f32_16x16x32_f16(
            /*neg_a=*/false, a, /*neg_b=*/false, b,
            /*c_mod=*/(short)0, acc, /*reuse_a=*/false, /*reuse_b=*/false);
    }
    // every row of D equals the column sum; take row M=0 from lanes 0..15
    if (lane < 16) atomicAdd(&me_sum[eb + lane], acc[0]);
    __syncthreads();

    float part = 0.0f;
    if (tid < N_EXP) part = me_sum[tid] * (float)cnt1[tid];
    for (int o = 16; o > 0; o >>= 1) part += __shfl_xor(part, o, 32);
    if (lane == 0 && wave < 2) red[wave] = part;
    __syncthreads();
    if (tid == 0)
        out_laux[0] = (float)N_EXP * (red[0] + red[1]) / ((float)S_TOK * (float)S_TOK);
}

// ---------------------------------------------------------------------------
// K4: fused zero-fill + scatter, b128 stores. One block per token owns its
// [E,C]=8192-float slab of combine_weights and dispatch_mask. The leading
// l_aux scalar shifts both arrays to byte-phase 4 (mod 16), so each slab is
// written as 3 head floats + 2047 aligned float4 chunks + 1 tail float.
// dispatch_mask == (combine != 0) exactly (reference: combine.astype(bool),
// and kept gate weights are strictly positive).
// ---------------------------------------------------------------------------
__global__ void write_kernel(const int* __restrict__ idx1, const int* __restrict__ idx2,
                             const int* __restrict__ loc1, const int* __restrict__ loc2,
                             const float* __restrict__ g1raw, const float* __restrict__ g2raw,
                             const int* __restrict__ cap_ptr,
                             float* __restrict__ combine, float* __restrict__ dmask) {
    const int s = blockIdx.x;
    const int tid = threadIdx.x;                           // blockDim = 256
    const int cap = cap_ptr[0];

    const int i1 = idx1[s], i2 = idx2[s];
    const int l1 = loc1[s], l2 = loc2[s];
    const bool k1 = l1 < cap, k2 = l2 < cap;
    const float ga = k1 ? g1raw[s] : 0.0f;
    const float gb = k2 ? g2raw[s] : 0.0f;
    const float denom = fmaxf(ga + gb, FLT_EPS);
    const float w1 = ga / denom, w2 = gb / denom;
    const int p1 = k1 ? (i1 * 128 + l1) : -1;              // slab index in [0, 8192)
    const int p2 = k2 ? (i2 * 128 + l2) : -1;

    float* __restrict__ crow = combine + (size_t)s * 8192; // byte phase 4 (mod 16)
    float* __restrict__ mrow = dmask   + (size_t)s * 8192; // same phase (offset mult. of 16B)

    // head (idx 0..2) and tail (idx 8191) scalars
    if (tid == 0) {
#pragma unroll
        for (int j = 0; j < 4; ++j) {
            const int idx = (j < 3) ? j : 8191;
            float cw = (idx == p1) ? w1 : ((idx == p2) ? w2 : 0.0f);
            crow[idx] = cw;
            mrow[idx] = (cw != 0.0f) ? 1.0f : 0.0f;
        }
    }

    // 2047 aligned float4 chunks covering idx 3 .. 8190
    for (int c = tid; c < 2047; c += 256) {
        const int base = 3 + c * 4;                        // byte-aligned to 16
        float4 cv, mv;
        cv.x = (base + 0 == p1) ? w1 : ((base + 0 == p2) ? w2 : 0.0f);
        cv.y = (base + 1 == p1) ? w1 : ((base + 1 == p2) ? w2 : 0.0f);
        cv.z = (base + 2 == p1) ? w1 : ((base + 2 == p2) ? w2 : 0.0f);
        cv.w = (base + 3 == p1) ? w1 : ((base + 3 == p2) ? w2 : 0.0f);
        mv.x = (cv.x != 0.0f) ? 1.0f : 0.0f;
        mv.y = (cv.y != 0.0f) ? 1.0f : 0.0f;
        mv.z = (cv.z != 0.0f) ? 1.0f : 0.0f;
        mv.w = (cv.w != 0.0f) ? 1.0f : 0.0f;
        *reinterpret_cast<float4*>(crow + base) = cv;
        *reinterpret_cast<float4*>(mrow + base) = mv;
    }
}

// ---------------------------------------------------------------------------
extern "C" void kernel_launch(void* const* d_in, const int* in_sizes, int n_in,
                              void* d_out, int out_size, void* d_ws, size_t ws_size,
                              hipStream_t stream) {
    (void)in_sizes; (void)n_in; (void)out_size; (void)ws_size;
    const float* logits = (const float*)d_in[0];
    const float* noise  = (const float*)d_in[1];
    const int*   cap    = (const int*)d_in[2];

    // workspace layout
    float* ws    = (float*)d_ws;
    float* gates = ws;                                   // S*E floats
    int*   idx1  = (int*)(ws + (size_t)S_TOK * N_EXP);   // S ints
    int*   idx2  = idx1 + S_TOK;
    int*   loc1  = idx2 + S_TOK;
    int*   loc2  = loc1 + S_TOK;
    float* g1    = (float*)(loc2 + S_TOK);
    float* g2    = g1 + S_TOK;
    int*   cnt1  = (int*)(g2 + S_TOK);                   // E ints

    // output layout: [l_aux][combine S*E*C][dispatch_mask S*E*C]
    float* outp    = (float*)d_out;
    float* laux    = outp;
    float* combine = outp + 1;
    float* dmask   = outp + 1 + (size_t)S_TOK * N_EXP * 128;

    gate_kernel<<<S_TOK / 8, 256, 0, stream>>>(logits, noise, gates, idx1, idx2, g1, g2);
    scan_kernel<<<N_EXP, 32, 0, stream>>>(idx1, idx2, loc1, loc2, cnt1);
    laux_kernel<<<1, 256, 0, stream>>>(gates, cnt1, laux);
    write_kernel<<<S_TOK, 256, 0, stream>>>(idx1, idx2, loc1, loc2, g1, g2, cap, combine, dmask);
}
